// PairwiseFairnessLoss_63230508532296
// MI455X (gfx1250) — matrix operand order repaired and checked by reading, not compiled, not run on hardware
//
#include <hip/hip_runtime.h>
#include <hip/hip_bf16.h>

// ---------------------------------------------------------------------------
// Debiased Sinkhorn divergence across 8 subgroups (geomloss-style) on gfx1250.
//  Phase 1: gather rows by subgroup (stable rank), split fp32 -> bf16 hi+lo,
//           half squared norms q = 0.5*||x||^2.
//  Phase 2: 36 Gram/cost matrices via v_wmma_f32_16x16x32_bf16, 3-pass bf16
//           split product (hi*hi + hi*lo + lo*hi ~ fp32 accuracy).
//           C (144 MB fp32) stays L2-resident on MI455X (192 MB L2).
//  Phase 3: ~115 eps-scaled Sinkhorn iterations: streaming logsumexp sweeps
//           (rows: wave-per-row, cols: thread-per-column + chunk merge).
//  Phase 4: final extrapolation at eps_target, pair losses + total.
// ---------------------------------------------------------------------------

typedef __attribute__((ext_vector_type(16))) __bf16       v16bf;
typedef __attribute__((ext_vector_type(8)))  float        v8f;
typedef __attribute__((ext_vector_type(4)))  unsigned int u32x4;
typedef __attribute__((ext_vector_type(8)))  unsigned int u32x8;

#define NSUB   8
#define NPS    1024          // samples per subgroup
#define DIM    256
#define NPAIR  28
#define NSLOT  36            // 28 cross + 8 self
#define LOGM   6.9314718055994531f   // ln(1024)

__constant__ int c_pi[NPAIR] = {0,0,0,0,0,0,0,1,1,1,1,1,1,2,2,2,2,2,3,3,3,3,4,4,4,5,5,6};
__constant__ int c_pj[NPAIR] = {1,2,3,4,5,6,7,2,3,4,5,6,7,3,4,5,6,7,4,5,6,7,5,6,7,6,7,7};

__device__ __forceinline__ unsigned short f32_to_bf16(float x) {
    unsigned u = __float_as_uint(x);
    unsigned r = u + 0x7FFFu + ((u >> 16) & 1u);   // round-to-nearest-even
    return (unsigned short)(r >> 16);
}
__device__ __forceinline__ float bf16_to_f32(unsigned short h) {
    return __uint_as_float(((unsigned)h) << 16);
}

// ------------------------------- Phase 1 -----------------------------------
__global__ __launch_bounds__(256)
void prep_kernel(const float* __restrict__ feat, const int* __restrict__ sub,
                 unsigned short* __restrict__ Ghi, unsigned short* __restrict__ Glo,
                 float* __restrict__ q) {
    const int r   = blockIdx.x;      // input row 0..8191
    const int tid = threadIdx.x;     // 0..255 (== feature dim)
    const int g   = sub[r];

    // stable rank of row r within its subgroup
    int cnt = 0;
    for (int rr = tid; rr < NSUB * NPS; rr += 256)
        if (rr < r && sub[rr] == g) cnt++;
    for (int o = 16; o > 0; o >>= 1) cnt += __shfl_xor(cnt, o, 32);
    __shared__ int ired[8];
    const int lane = tid & 31, wv = tid >> 5;
    if (lane == 0) ired[wv] = cnt;
    __syncthreads();
    __shared__ int rank_sh;
    if (tid == 0) {
        int s = 0;
        for (int k = 0; k < 8; k++) s += ired[k];
        rank_sh = s;
    }
    __syncthreads();
    const int pos = g * NPS + rank_sh;

    const float x  = feat[(size_t)r * DIM + tid];
    const unsigned short hb = f32_to_bf16(x);
    const unsigned short lb = f32_to_bf16(x - bf16_to_f32(hb));
    Ghi[(size_t)pos * DIM + tid] = hb;
    Glo[(size_t)pos * DIM + tid] = lb;

    float v = 0.5f * x * x;
    for (int o = 16; o > 0; o >>= 1) v += __shfl_xor(v, o, 32);
    __shared__ float fred[8];
    if (lane == 0) fred[wv] = v;
    __syncthreads();
    if (tid == 0) {
        float s = 0.f;
        for (int k = 0; k < 8; k++) s += fred[k];
        q[pos] = s;
    }
}

__global__ void zero_kernel(float* __restrict__ p, int n) {
    int i = blockIdx.x * blockDim.x + threadIdx.x;
    if (i < n) p[i] = 0.f;
}

// ------------------------------- Phase 2 -----------------------------------
// A fragment (16x32 bf16): lane l -> row M=l%16; two 16B chunks at K = hi*8
// and K = 16 + hi*8 (hi = l/16).  [ISA 7.12.2, 16-bit A table]
__device__ __forceinline__ v16bf load_a_frag(const unsigned short* rowbase, int kb, int hi) {
    u32x4 c0 = *(const u32x4*)(rowbase + kb + hi * 8);
    u32x4 c1 = *(const u32x4*)(rowbase + kb + 16 + hi * 8);
    u32x8 c  = __builtin_shufflevector(c0, c1, 0, 1, 2, 3, 4, 5, 6, 7);
    return __builtin_bit_cast(v16bf, c);
}
// B fragment (32x16 bf16): lane l -> col N=l%16; two 16B chunks at K = hi*8 and K = 16 + hi*8.
__device__ __forceinline__ v16bf load_b_frag(const unsigned short* rowbase, int kb, int hi) {
    u32x4 c0 = *(const u32x4*)(rowbase + kb + hi * 8);
    u32x4 c1 = *(const u32x4*)(rowbase + kb + 16 + hi * 8);
    u32x8 c  = __builtin_shufflevector(c0, c1, 0, 1, 2, 3, 4, 5, 6, 7);
    return __builtin_bit_cast(v16bf, c);
}

// block = 256 threads = 8 waves; each wave: 16 rows x 64 cols of one slot.
// grid = (16 n-tiles, 8 m-tiles, 36 slots)
__global__ __launch_bounds__(256)
void cost_kernel(const unsigned short* __restrict__ Ghi,
                 const unsigned short* __restrict__ Glo,
                 const float* __restrict__ q, float* __restrict__ C) {
    const int slot = blockIdx.z;
    int si, sj;
    if (slot < NPAIR) { si = c_pi[slot]; sj = c_pj[slot]; }
    else              { si = sj = slot - NPAIR; }

    const int tid  = threadIdx.x;
    const int wv   = tid >> 5;
    const int lane = tid & 31;
    const int hi   = lane >> 4;
    const int ln   = lane & 15;
    const int m0   = blockIdx.y * 128 + wv * 16;
    const int n0   = blockIdx.x * 64;

    const unsigned short* Ah = Ghi + ((size_t)(si * NPS + m0 + ln)) * DIM;
    const unsigned short* Al = Glo + ((size_t)(si * NPS + m0 + ln)) * DIM;
    const unsigned short* Bh = Ghi + ((size_t)(sj * NPS + n0 + ln)) * DIM;
    const unsigned short* Bl = Glo + ((size_t)(sj * NPS + n0 + ln)) * DIM;

    v8f acc[4];
    v8f zero = {0.f, 0.f, 0.f, 0.f, 0.f, 0.f, 0.f, 0.f};
#pragma unroll
    for (int t = 0; t < 4; t++) acc[t] = zero;

    for (int kb = 0; kb < DIM; kb += 32) {
        v16bf ahi = load_a_frag(Ah, kb, hi);
        v16bf alo = load_a_frag(Al, kb, hi);
#pragma unroll
        for (int t = 0; t < 4; t++) {
            const size_t roff = (size_t)t * 16 * DIM;
            v16bf bhi = load_b_frag(Bh + roff, kb, hi);
            v16bf blo = load_b_frag(Bl + roff, kb, hi);
            // 3-pass bf16 split product: hi*hi + hi*lo + lo*hi ~ fp32 dot
            acc[t] = __builtin_amdgcn_wmma_f32_16x16x32_bf16(false, ahi, false, bhi,
                                                             (short)0, acc[t], false, false);
            acc[t] = __builtin_amdgcn_wmma_f32_16x16x32_bf16(false, ahi, false, blo,
                                                             (short)0, acc[t], false, false);
            acc[t] = __builtin_amdgcn_wmma_f32_16x16x32_bf16(false, alo, false, bhi,
                                                             (short)0, acc[t], false, false);
        }
    }

    // C_ij = max(0.5||x_i||^2 + 0.5||y_j||^2 - x_i.y_j, 0)
    float* Cs = C + ((size_t)slot << 20);
    const float* qx = q + si * NPS;
    const float* qy = q + sj * NPS;
#pragma unroll
    for (int t = 0; t < 4; t++) {
        const int n   = n0 + t * 16 + ln;
        const float b = qy[n];
#pragma unroll
        for (int r = 0; r < 8; r++) {
            const int m = m0 + r + 8 * hi;          // D layout: M = r + 8*(lane/16)
            const float c = qx[m] + b - acc[t][r];
            Cs[((size_t)m << 10) + n] = fmaxf(c, 0.f);
        }
    }
}

// ------------------------------- Phase 3 -----------------------------------
// Row softmin: out[p][i] = -eps*(LSE_j((h[p][j]-C[p][i][j])/eps) - ln m)
// one wave per row. mode==1: out = 0.5*(old[row] + softmin)   (self update)
__global__ __launch_bounds__(256)
void softmin_rows(const float* __restrict__ C, const float* __restrict__ h,
                  const float* __restrict__ oldv, float* __restrict__ out,
                  float eps, int nmat, int mode) {
    const int gw   = blockIdx.x * 8 + (threadIdx.x >> 5);
    const int lane = threadIdx.x & 31;
    const int p    = gw >> 10;
    const int i    = gw & 1023;
    if (p >= nmat) return;

    const float* Crow = C + ((size_t)p << 20) + ((size_t)i << 10);
    const float* hp   = h + (p << 10);
    const float  inv  = 1.0f / eps;

    float M = -3.4e38f, S = 0.f;
    for (int j = lane; j < NPS; j += 32) {
        const float t = (hp[j] - Crow[j]) * inv;
        if (t > M) { S = S * __expf(M - t) + 1.f; M = t; }
        else       { S += __expf(t - M); }
    }
    for (int o = 16; o > 0; o >>= 1) {
        const float Mo = __shfl_xor(M, o, 32);
        const float So = __shfl_xor(S, o, 32);
        const float Mn = fmaxf(M, Mo);
        S = S * __expf(M - Mn) + So * __expf(Mo - Mn);
        M = Mn;
    }
    if (lane == 0) {
        float v = -eps * (M + __logf(S) - LOGM);
        if (mode == 1) v = 0.5f * (oldv[gw] + v);
        out[gw] = v;
    }
}

// Column softmin, pass 1: partial streaming LSE over a 256-row chunk.
// grid = (4 col-chunks, 4 row-chunks, 28 pairs), thread = one column.
__global__ __launch_bounds__(256)
void softmin_cols_partial(const float* __restrict__ C, const float* __restrict__ h,
                          float* __restrict__ pM, float* __restrict__ pS, float eps) {
    const int p  = blockIdx.z;
    const int j  = blockIdx.x * 256 + threadIdx.x;
    const int i0 = blockIdx.y * 256;

    __shared__ float hs[256];
    hs[threadIdx.x] = h[(p << 10) + i0 + threadIdx.x];
    __syncthreads();

    const float* Cp  = C + ((size_t)p << 20);
    const float  inv = 1.0f / eps;
    float M = -3.4e38f, S = 0.f;
    for (int ii = 0; ii < 256; ++ii) {
        const float t = (hs[ii] - Cp[((size_t)(i0 + ii) << 10) + j]) * inv;
        if (t > M) { S = S * __expf(M - t) + 1.f; M = t; }
        else       { S += __expf(t - M); }
    }
    const int idx = ((p * 4 + blockIdx.y) << 10) + j;
    pM[idx] = M;
    pS[idx] = S;
}

// Column softmin, pass 2: merge 4 chunk (max,sum) pairs.
// grid = (4 col-chunks, 28 pairs)
__global__ __launch_bounds__(256)
void softmin_cols_combine(const float* __restrict__ pM, const float* __restrict__ pS,
                          float* __restrict__ out, float eps) {
    const int p = blockIdx.y;
    const int j = blockIdx.x * 256 + threadIdx.x;
    float M = -3.4e38f;
#pragma unroll
    for (int c = 0; c < 4; c++) M = fmaxf(M, pM[((p * 4 + c) << 10) + j]);
    float S = 0.f;
#pragma unroll
    for (int c = 0; c < 4; c++) S += pS[((p * 4 + c) << 10) + j] * __expf(pM[((p * 4 + c) << 10) + j] - M);
    out[(p << 10) + j] = -eps * (M + __logf(S) - LOGM);
}

// ------------------------------- Phase 4 -----------------------------------
__global__ __launch_bounds__(256)
void pair_loss_kernel(const float* __restrict__ ff, const float* __restrict__ gf,
                      const float* __restrict__ sf, float* __restrict__ out) {
    const int p  = blockIdx.x;
    const int si = c_pi[p], sj = c_pj[p];
    float a = 0.f;
    for (int i = threadIdx.x; i < NPS; i += 256)
        a += (ff[(p << 10) + i] - sf[(si << 10) + i]) +
             (gf[(p << 10) + i] - sf[(sj << 10) + i]);
    for (int o = 16; o > 0; o >>= 1) a += __shfl_xor(a, o, 32);
    __shared__ float red[8];
    const int lane = threadIdx.x & 31, wv = threadIdx.x >> 5;
    if (lane == 0) red[wv] = a;
    __syncthreads();
    if (threadIdx.x == 0) {
        float s = 0.f;
        for (int k = 0; k < 8; k++) s += red[k];
        out[1 + p] = s * (1.0f / NPS);
    }
}

__global__ void total_kernel(float* __restrict__ out) {
    float v = (threadIdx.x < NPAIR) ? out[1 + threadIdx.x] : 0.f;
    for (int o = 16; o > 0; o >>= 1) v += __shfl_xor(v, o, 32);
    if (threadIdx.x == 0) out[0] = v * (1.0f / NPAIR);
}

// ---------------------------------------------------------------------------
extern "C" void kernel_launch(void* const* d_in, const int* in_sizes, int n_in,
                              void* d_out, int out_size, void* d_ws, size_t ws_size,
                              hipStream_t stream) {
    (void)in_sizes; (void)n_in; (void)out_size; (void)ws_size;
    const float* feat = (const float*)d_in[0];
    const int*   sub  = (const int*)d_in[1];
    float*       out  = (float*)d_out;

    // ---- workspace layout (~153.5 MiB) ----
    char* w = (char*)d_ws;
    size_t off = 0;
    unsigned short* Ghi = (unsigned short*)(w + off); off += (size_t)NSUB * NPS * DIM * 2;
    unsigned short* Glo = (unsigned short*)(w + off); off += (size_t)NSUB * NPS * DIM * 2;
    float* q   = (float*)(w + off); off += (size_t)NSUB * NPS * 4;
    float* C   = (float*)(w + off); off += (size_t)NSLOT * NPS * NPS * 4;   // 144 MiB
    float* Cxx = C + ((size_t)NPAIR << 20);
    const size_t FG = (size_t)NPAIR * NPS;   // 28672
    const size_t SS = (size_t)NSUB * NPS;    // 8192
    float* dual0 = (float*)(w + off); off += (2 * FG + SS) * 4;   // f0,g0,s0 contiguous
    float* dual1 = (float*)(w + off); off += (2 * FG + SS) * 4;
    float* fb[2] = { dual0,            dual1 };
    float* gb[2] = { dual0 + FG,       dual1 + FG };
    float* sb[2] = { dual0 + 2 * FG,   dual1 + 2 * FG };
    float* f_fin = (float*)(w + off); off += FG * 4;
    float* g_fin = (float*)(w + off); off += FG * 4;
    float* s_fin = (float*)(w + off); off += SS * 4;
    float* pM    = (float*)(w + off); off += (size_t)NPAIR * 4 * NPS * 4;
    float* pS    = (float*)(w + off); off += (size_t)NPAIR * 4 * NPS * 4;

    // ---- phase 1: gather + bf16 split + norms, zero duals ----
    prep_kernel<<<NSUB * NPS, 256, 0, stream>>>(feat, sub, Ghi, Glo, q);
    {
        const int nz = (int)(2 * FG + SS);
        zero_kernel<<<(nz + 255) / 256, 256, 0, stream>>>(dual0, nz);
    }

    // ---- phase 2: 36 cost matrices via WMMA ----
    cost_kernel<<<dim3(16, 8, NSLOT), 256, 0, stream>>>(Ghi, Glo, q, C);

    // ---- phase 3: eps-scaled Sinkhorn loop (static schedule) ----
    const int rows_blk_cross = NPAIR * NPS / 8;   // 3584
    const int rows_blk_self  = NSUB * NPS / 8;    // 1024
    int cur = 0;
    double eps = (double)DIM;                     // eps0 = D = 256
    const double target = 1e-8;                   // blur^2
    bool done = false;
    while (!done) {
        float e;
        if (eps > target) { e = (float)eps; eps *= 0.81; }   // ratio = scaling^2
        else              { e = (float)target; done = true; }
        // f_t = softmin(eps, Cxy, g)
        softmin_rows<<<rows_blk_cross, 256, 0, stream>>>(C, gb[cur], nullptr, fb[1 - cur], e, NPAIR, 0);
        // g_t = softmin(eps, Cxy^T, f)  (column sweep)
        softmin_cols_partial<<<dim3(4, 4, NPAIR), 256, 0, stream>>>(C, fb[cur], pM, pS, e);
        softmin_cols_combine<<<dim3(4, NPAIR), 256, 0, stream>>>(pM, pS, gb[1 - cur], e);
        // s_t = 0.5*(s + softmin(eps, Cxx, s))
        softmin_rows<<<rows_blk_self, 256, 0, stream>>>(Cxx, sb[cur], sb[cur], sb[1 - cur], e, NSUB, 1);
        cur ^= 1;
    }

    // ---- phase 4: final extrapolation at eps_target + losses ----
    const float et = (float)target;
    softmin_rows<<<rows_blk_cross, 256, 0, stream>>>(C, gb[cur], nullptr, f_fin, et, NPAIR, 0);
    softmin_cols_partial<<<dim3(4, 4, NPAIR), 256, 0, stream>>>(C, fb[cur], pM, pS, et);
    softmin_cols_combine<<<dim3(4, NPAIR), 256, 0, stream>>>(pM, pS, g_fin, et);
    softmin_rows<<<rows_blk_self, 256, 0, stream>>>(Cxx, sb[cur], nullptr, s_fin, et, NSUB, 0);

    pair_loss_kernel<<<NPAIR, 256, 0, stream>>>(f_fin, g_fin, s_fin, out);
    total_kernel<<<1, 32, 0, stream>>>(out);
}
